// Seq2Seq_14147622273712
// MI455X (gfx1250) — matrix operand order repaired and checked by reading, not compiled, not run on hardware
//
#include <hip/hip_runtime.h>
#include <hip/hip_bf16.h>
#include <math.h>

// ---------------------------------------------------------------------------
// Seq2Seq LSTM encoder/decoder for MI455X (gfx1250).
//  * All big matvecs run through v_wmma_f32_16x16x32_bf16 with fp32 accum.
//  * Weights converted once per launch fp32->bf16 into workspace: 99 MB total,
//    fully L2-resident (192 MB L2) -> each of the 512 sequential steps streams
//    weights from L2, not HBM.
//  * fc phase processes tile PAIRS so each B (vector) load feeds two WMMAs
//    on independent accumulator chains; inner loops capped at unroll 2 to
//    prevent the hoist-all-B-vectors -> scratch-spill pathology.
//  * Single persistent cooperative kernel; software grid barrier between the
//    phases of each time step. Deterministic (fixed reduction trees, u64
//    atomicMax for argmax).
//  * Workspace requirement: ~95 MiB.
// ---------------------------------------------------------------------------

#define HID     1024
#define GATES   4096
#define VOUT    32000
#define SLEN    256
#define TLEN    256
#define NB      64
#define TPB     512
#define NWAVES  ((NB * TPB) / 32)   // 1024 waves
#define GRIDTH  (NB * TPB)          // 32768 threads
#define VSLICE  (VOUT / NB)         // 500 logits per block for the exp-sum

typedef __attribute__((ext_vector_type(16))) __bf16 v16bf;
typedef __attribute__((ext_vector_type(8)))  __bf16 v8bf;
typedef __attribute__((ext_vector_type(8)))  float  v8f;

// ---- workspace layout (bytes) ----
#define OFF_CTRL    0                         // 256 B control block
#define OFF_H       256
#define OFF_C       (OFF_H + HID * 4)
#define OFF_HB      (OFF_C + HID * 4)         // bf16 hidden state
#define OFF_XB      (OFF_HB + HID * 2)        // bf16 current input embedding
#define OFF_G       (OFF_XB + HID * 2)        // fp32 gate pre-activations [4096]
#define OFF_PSUM    (OFF_G + GATES * 4)       // per-block exp-sum partials [64]
#define OFF_LOGITS  (OFF_PSUM + NB * 4)       // fp32 logits [32000]
#define OFF_WB      (OFF_LOGITS + VOUT * 4)   // bf16 weights start (32B aligned)
// bf16 weight element offsets inside WB region
#define EW_ENC_WIH  ((size_t)0)
#define EW_ENC_WHH  ((size_t)4 * HID * HID)
#define EW_DEC_WIH  ((size_t)8 * HID * HID)
#define EW_DEC_WHH  ((size_t)12 * HID * HID)
#define EW_FC_W     ((size_t)16 * HID * HID)

// ---------------- helpers ----------------

__device__ __forceinline__ unsigned short f2bf(float f) {
    unsigned u = __float_as_uint(f);
    unsigned r = ((u >> 16) & 1u) + 0x7FFFu;     // round to nearest even
    return (unsigned short)((u + r) >> 16);
}

__device__ __forceinline__ unsigned map_ord(float f) {
    unsigned b = __float_as_uint(f);
    return (b & 0x80000000u) ? ~b : (b | 0x80000000u);   // monotone float->uint
}
__device__ __forceinline__ float unmap_ord(unsigned m) {
    unsigned b = (m & 0x80000000u) ? (m & 0x7FFFFFFFu) : ~m;
    return __uint_as_float(b);
}

__device__ __forceinline__ void grid_sync(unsigned* cnt, unsigned* gen, unsigned nb) {
    __syncthreads();
    if (threadIdx.x == 0) {
        __threadfence();
        unsigned old = __hip_atomic_load(gen, __ATOMIC_RELAXED, __HIP_MEMORY_SCOPE_AGENT);
        unsigned arrived = __hip_atomic_fetch_add(cnt, 1u, __ATOMIC_ACQ_REL, __HIP_MEMORY_SCOPE_AGENT);
        if (arrived == nb - 1u) {
            __hip_atomic_store(cnt, 0u, __ATOMIC_RELAXED, __HIP_MEMORY_SCOPE_AGENT);
            __hip_atomic_fetch_add(gen, 1u, __ATOMIC_RELEASE, __HIP_MEMORY_SCOPE_AGENT);
        } else {
            while (__hip_atomic_load(gen, __ATOMIC_ACQUIRE, __HIP_MEMORY_SCOPE_AGENT) == old) {
                __builtin_amdgcn_s_sleep(2);
            }
        }
        __threadfence();
    }
    __syncthreads();
}

// Deterministic fixed-tree block reduction (wave32). Result valid in thread 0.
__device__ __forceinline__ float block_reduce_sum(float v) {
    __shared__ float red[TPB / 32];
    #pragma unroll
    for (int m = 16; m > 0; m >>= 1) v += __shfl_xor(v, m, 32);
    int w = threadIdx.x >> 5;
    if ((threadIdx.x & 31) == 0) red[w] = v;
    __syncthreads();
    if (w == 0) {
        v = (threadIdx.x < (TPB / 32)) ? red[threadIdx.x] : 0.0f;
        #pragma unroll
        for (int m = 8; m > 0; m >>= 1) v += __shfl_xor(v, m, 32);
    }
    __syncthreads();
    return v;
}

// Build the A fragment (16 rows x 32 K weight tile) from two 16B loads.
__device__ __forceinline__ v16bf make_a(const __bf16* wr, int kb) {
    v8bf t0 = *(const v8bf*)(wr + kb);        // K = kb+off .. +7
    v8bf t1 = *(const v8bf*)(wr + kb + 16);   // K = kb+off+16 .. +23
    v16bf a;
    #pragma unroll
    for (int i = 0; i < 8; ++i) { a[i] = t0[i]; a[8 + i] = t1[i]; }
    return a;
}

// Fused LSTM gate matvec for one 16-row tile:
//   y[mb..mb+15] = Wx[tile]@x + Wh[tile]@h
// A = weight tile, B = vector chunk replicated across 16 columns.
// After the loop lanes 0-15 hold y[mb+r] in acc[r], lanes 16-31 y[mb+8+r].
__device__ __forceinline__ v8f mv16_gates(const __bf16* __restrict__ Wx,
                                          const __bf16* __restrict__ Wh,
                                          const __bf16* __restrict__ x,
                                          const __bf16* __restrict__ h,
                                          int lane) {
    const int r  = lane & 15;
    const int ko = (lane & 16) ? 8 : 0;
    const __bf16* wx = Wx + (size_t)r * HID + ko;
    const __bf16* wh = Wh + (size_t)r * HID + ko;
    const __bf16* xp = x + ko;
    const __bf16* hp = h + ko;
    v8f acc = {};
    #pragma unroll 2
    for (int kb = 0; kb < HID; kb += 32) {
        v16bf bx = make_a(xp, kb);
        v16bf bh = make_a(hp, kb);
        v16bf ax = make_a(wx, kb);
        v16bf ah = make_a(wh, kb);
        acc = __builtin_amdgcn_wmma_f32_16x16x32_bf16(false, ax, false, bx,
                                                      (short)0, acc, false, false);
        acc = __builtin_amdgcn_wmma_f32_16x16x32_bf16(false, ah, false, bh,
                                                      (short)0, acc, false, false);
    }
    return acc;
}

// Pair of 16-row tiles sharing one B stream (halves vector traffic, two
// independent WMMA accumulation chains).
__device__ __forceinline__ void mv16_pair(v8f& acc0, v8f& acc1,
                                          const __bf16* __restrict__ W0,
                                          const __bf16* __restrict__ W1,
                                          const __bf16* __restrict__ x,
                                          int lane) {
    const int r  = lane & 15;
    const int ko = (lane & 16) ? 8 : 0;
    const __bf16* w0 = W0 + (size_t)r * HID + ko;
    const __bf16* w1 = W1 + (size_t)r * HID + ko;
    const __bf16* xp = x + ko;
    #pragma unroll 2
    for (int kb = 0; kb < HID; kb += 32) {
        v16bf b  = make_a(xp, kb);
        v16bf a0 = make_a(w0, kb);
        v16bf a1 = make_a(w1, kb);
        acc0 = __builtin_amdgcn_wmma_f32_16x16x32_bf16(false, a0, false, b,
                                                       (short)0, acc0, false, false);
        acc1 = __builtin_amdgcn_wmma_f32_16x16x32_bf16(false, a1, false, b,
                                                       (short)0, acc1, false, false);
    }
}

// ---------------- kernels ----------------

__global__ void cvt_f32_bf16(const float* __restrict__ s,
                             unsigned short* __restrict__ d, long n) {
    long i  = (long)blockIdx.x * blockDim.x + threadIdx.x;
    long st = (long)gridDim.x * blockDim.x;
    for (; i < n; i += st) d[i] = f2bf(s[i]);
}

__global__ __launch_bounds__(TPB)
void seq2seq_persistent(const int* __restrict__ src,
                        const float* __restrict__ enc_emb,
                        const float* __restrict__ enc_bih,
                        const float* __restrict__ enc_bhh,
                        const float* __restrict__ dec_emb,
                        const float* __restrict__ dec_bih,
                        const float* __restrict__ dec_bhh,
                        const float* __restrict__ fc_b,
                        char* __restrict__ wsb,
                        float* __restrict__ out) {
    unsigned* cnt  = (unsigned*)(wsb + OFF_CTRL);
    unsigned* gen  = cnt + 1;
    unsigned long long* gkey = (unsigned long long*)(wsb + 8);
    volatile unsigned* tokp  = (volatile unsigned*)(wsb + 16);
    volatile float*    logZp = (volatile float*)(wsb + 20);
    float* h      = (float*)(wsb + OFF_H);
    float* c      = (float*)(wsb + OFF_C);
    unsigned short* hb_u = (unsigned short*)(wsb + OFF_HB);
    unsigned short* xb_u = (unsigned short*)(wsb + OFF_XB);
    const __bf16* hb = (const __bf16*)hb_u;
    const __bf16* xb = (const __bf16*)xb_u;
    float* g      = (float*)(wsb + OFF_G);
    float* psum   = (float*)(wsb + OFF_PSUM);
    float* logits = (float*)(wsb + OFF_LOGITS);
    const __bf16* wb     = (const __bf16*)(wsb + OFF_WB);
    const __bf16* encWih = wb + EW_ENC_WIH;
    const __bf16* encWhh = wb + EW_ENC_WHH;
    const __bf16* decWih = wb + EW_DEC_WIH;
    const __bf16* decWhh = wb + EW_DEC_WHH;
    const __bf16* fcW    = wb + EW_FC_W;

    const int tid  = threadIdx.x;
    const int bid  = blockIdx.x;
    const int gtid = bid * TPB + tid;
    const int lane = tid & 31;
    const int wv   = gtid >> 5;
    const unsigned nb = gridDim.x;

    // ---- init: x0 = enc_emb[src[0]] (h, c, hb zeroed by host-side memset) ----
    if (gtid < HID) xb_u[gtid] = f2bf(enc_emb[(size_t)src[0] * HID + gtid]);
    grid_sync(cnt, gen, nb);

    // =================== encoder ===================
    for (int t = 0; t < SLEN; ++t) {
        // ---- gates: 256 tiles of 16 rows, one per wave ----
        for (int tile = wv; tile < GATES / 16; tile += NWAVES) {
            int mb = tile << 4;
            v8f acc = mv16_gates(encWih + ((size_t)mb * HID),
                                 encWhh + ((size_t)mb * HID), xb, hb, lane);
            if ((lane & 15) == 0) {
                int base = mb + ((lane & 16) ? 8 : 0);
                #pragma unroll
                for (int j = 0; j < 8; ++j) g[base + j] = acc[j];
            }
        }
        grid_sync(cnt, gen, nb);

        if (gtid < HID) {
            int i = gtid;
            float gi = g[i]           + enc_bih[i]           + enc_bhh[i];
            float gf = g[HID + i]     + enc_bih[HID + i]     + enc_bhh[HID + i];
            float gc = g[2 * HID + i] + enc_bih[2 * HID + i] + enc_bhh[2 * HID + i];
            float go = g[3 * HID + i] + enc_bih[3 * HID + i] + enc_bhh[3 * HID + i];
            float si = 1.0f / (1.0f + expf(-gi));
            float sf = 1.0f / (1.0f + expf(-gf));
            float tg = tanhf(gc);
            float so = 1.0f / (1.0f + expf(-go));
            float cn = sf * c[i] + si * tg;
            float hn = so * tanhf(cn);
            c[i] = cn; h[i] = hn; hb_u[i] = f2bf(hn);
            if (t + 1 < SLEN) xb_u[i] = f2bf(enc_emb[(size_t)src[t + 1] * HID + i]);
        }
        grid_sync(cnt, gen, nb);
    }

    // ---- transition: x = dec_emb[SOS=0] ----
    if (gtid < HID) xb_u[gtid] = f2bf(dec_emb[gtid]);
    grid_sync(cnt, gen, nb);

    // =================== decoder ===================
    for (int t = 0; t < TLEN; ++t) {
        // ---- phase A: write logp of step t-1 (logits stable) + gate matvecs ----
        if (t > 0) {
            float logZ = *logZp;
            for (int v = gtid; v < VOUT; v += GRIDTH)
                out[(size_t)(t - 1) * VOUT + v] = logits[v] - logZ;
        }
        for (int tile = wv; tile < GATES / 16; tile += NWAVES) {
            int mb = tile << 4;
            v8f acc = mv16_gates(decWih + ((size_t)mb * HID),
                                 decWhh + ((size_t)mb * HID), xb, hb, lane);
            if ((lane & 15) == 0) {
                int base = mb + ((lane & 16) ? 8 : 0);
                #pragma unroll
                for (int j = 0; j < 8; ++j) g[base + j] = acc[j];
            }
        }
        grid_sync(cnt, gen, nb);

        // ---- phase B: cell update + reset argmax key ----
        if (gtid < HID) {
            int i = gtid;
            float gi = g[i]           + dec_bih[i]           + dec_bhh[i];
            float gf = g[HID + i]     + dec_bih[HID + i]     + dec_bhh[HID + i];
            float gc = g[2 * HID + i] + dec_bih[2 * HID + i] + dec_bhh[2 * HID + i];
            float go = g[3 * HID + i] + dec_bih[3 * HID + i] + dec_bhh[3 * HID + i];
            float si = 1.0f / (1.0f + expf(-gi));
            float sf = 1.0f / (1.0f + expf(-gf));
            float tg = tanhf(gc);
            float so = 1.0f / (1.0f + expf(-go));
            float cn = sf * c[i] + si * tg;
            float hn = so * tanhf(cn);
            c[i] = cn; h[i] = hn; hb_u[i] = f2bf(hn);
        }
        if (gtid == 0) *gkey = 0ull;
        grid_sync(cnt, gen, nb);

        // ---- phase C: logits = fcW@h + fc_b, tile pairs (B reuse x2),
        //      plus global (max, argmax) key via order-independent atomicMax ----
        for (int p = wv; p < (VOUT / 32); p += NWAVES) {   // 1000 pairs
            int vb = p << 5;                                // rows vb .. vb+31
            v8f acc0 = {}, acc1 = {};
            mv16_pair(acc0, acc1,
                      fcW + ((size_t)vb * HID),
                      fcW + ((size_t)(vb + 16) * HID), hb, lane);
            if ((lane & 15) == 0) {
                int b0 = vb + ((lane & 16) ? 8 : 0);
                int b1 = b0 + 16;
                float lmax = -3.402823466e38f;
                unsigned lidx = 0;
                #pragma unroll
                for (int j = 0; j < 8; ++j) {
                    float l0 = acc0[j] + fc_b[b0 + j];
                    float l1 = acc1[j] + fc_b[b1 + j];
                    logits[b0 + j] = l0;
                    logits[b1 + j] = l1;
                    if (l0 > lmax) { lmax = l0; lidx = (unsigned)(b0 + j); }
                    if (l1 > lmax) { lmax = l1; lidx = (unsigned)(b1 + j); }
                }
                unsigned long long key =
                    ((unsigned long long)map_ord(lmax) << 32) |
                    (unsigned long long)(0xFFFFFFFFu - lidx);   // ties -> lowest idx
                atomicMax(gkey, key);
            }
        }
        grid_sync(cnt, gen, nb);

        // ---- phase E: per-block partial sum of exp(l - gmax), fixed tree ----
        {
            unsigned long long key = *(volatile unsigned long long*)gkey;
            float gmax = unmap_ord((unsigned)(key >> 32));
            float s = 0.0f;
            int lo = bid * VSLICE, hiend = lo + VSLICE;
            for (int v = lo + tid; v < hiend; v += TPB) s += expf(logits[v] - gmax);
            float S = block_reduce_sum(s);
            if (tid == 0) psum[bid] = S;
        }
        grid_sync(cnt, gen, nb);

        // ---- phase F (block 0): logZ, argmax token, next embedding -> xb ----
        if (bid == 0) {
            float v = (tid < NB) ? psum[tid] : 0.0f;
            float S = block_reduce_sum(v);
            unsigned long long key = *(volatile unsigned long long*)gkey;
            if (tid == 0) {
                float gmax = unmap_ord((unsigned)(key >> 32));
                *logZp = gmax + logf(S);
                *tokp  = 0xFFFFFFFFu - (unsigned)(key & 0xFFFFFFFFu);
            }
            __syncthreads();
            unsigned tok = *tokp;
            for (int i = tid; i < HID; i += TPB)
                xb_u[i] = f2bf(dec_emb[(size_t)tok * HID + i]);
        }
        grid_sync(cnt, gen, nb);
    }

    // ---- final logp write for step TLEN-1 ----
    {
        float logZ = *logZp;
        for (int v = gtid; v < VOUT; v += GRIDTH)
            out[(size_t)(TLEN - 1) * VOUT + v] = logits[v] - logZ;
    }
}

// ---------------- launch ----------------

extern "C" void kernel_launch(void* const* d_in, const int* in_sizes, int n_in,
                              void* d_out, int out_size, void* d_ws, size_t ws_size,
                              hipStream_t stream) {
    (void)in_sizes; (void)n_in; (void)out_size; (void)ws_size;
    const int*   src     = (const int*)d_in[0];
    const float* enc_emb = (const float*)d_in[2];
    const float* enc_Wih = (const float*)d_in[3];
    const float* enc_Whh = (const float*)d_in[4];
    const float* enc_bih = (const float*)d_in[5];
    const float* enc_bhh = (const float*)d_in[6];
    const float* dec_emb = (const float*)d_in[7];
    const float* dec_Wih = (const float*)d_in[8];
    const float* dec_Whh = (const float*)d_in[9];
    const float* dec_bih = (const float*)d_in[10];
    const float* dec_bhh = (const float*)d_in[11];
    const float* fc_W    = (const float*)d_in[12];
    const float* fc_b    = (const float*)d_in[13];
    float* out = (float*)d_out;
    char* wsb = (char*)d_ws;
    unsigned short* wbu = (unsigned short*)(wsb + OFF_WB);

    // zero control block + h + c + hb (bf16 zero == 0x0000)
    hipMemsetAsync(wsb, 0, OFF_XB, stream);

    const long nG = 4L * HID * HID;
    cvt_f32_bf16<<<2048, 256, 0, stream>>>(enc_Wih, wbu + EW_ENC_WIH, nG);
    cvt_f32_bf16<<<2048, 256, 0, stream>>>(enc_Whh, wbu + EW_ENC_WHH, nG);
    cvt_f32_bf16<<<2048, 256, 0, stream>>>(dec_Wih, wbu + EW_DEC_WIH, nG);
    cvt_f32_bf16<<<2048, 256, 0, stream>>>(dec_Whh, wbu + EW_DEC_WHH, nG);
    cvt_f32_bf16<<<2048, 256, 0, stream>>>(fc_W,    wbu + EW_FC_W, (long)VOUT * HID);

    seq2seq_persistent<<<NB, TPB, 0, stream>>>(
        src, enc_emb, enc_bih, enc_bhh, dec_emb, dec_bih, dec_bhh, fc_b, wsb, out);
}
